// EdgeNetwork_24635932410274
// MI455X (gfx1250) — matrix operand order repaired and checked by reading, not compiled, not run on hardware
//
#include <hip/hip_runtime.h>

typedef __attribute__((ext_vector_type(16))) _Float16 v16h;
typedef __attribute__((ext_vector_type(8)))  float    v8f;

#define D            32          // atom feature dim
#define BD           16          // bond feature dim
#define NSTEP        17          // K-steps of 32: 16 (bond x neigh) + 1 (bias x neigh)
#define HALF_ELEMS   (NSTEP * 2 * 32 * 16)   // 17408 f16 per precision level
#define WG_THREADS   128
#define WAVES_PER_WG 4
#define BOND_PITCH   20          // floats per staged bond row (conflict-free padding)

// --------------------------------------------------------------------------
// Prep: build Waug (544 x 32) directly in the WMMA B-operand lane layout.
//   Waug[k*32 + j, i] = W[k, i*32 + j]   (rows 0..511)
//   Waug[512 + j,  i] = b[i*32 + j]      (rows 512..543, bias folded into GEMM)
// B order used here (16-bit, 32x16, wave32; contiguous K, same as A): lane L holds column N = L%16,
//   K = t + 16*(L>>4) for half-index t (K=2v,2v+1 per VGPR v).
// f16 hi at ws[q], f16 residual lo at ws[HALF_ELEMS + q].
// --------------------------------------------------------------------------
__global__ void prep_waug(const float* __restrict__ W, const float* __restrict__ bias,
                          _Float16* __restrict__ ws) {
  int q = blockIdx.x * blockDim.x + threadIdx.x;
  if (q >= HALF_ELEMS) return;
  int t    = q & 15;
  int lane = (q >> 4) & 31;
  int sh   = q >> 9;                    // 0..33 = s*2 + h
  int h = sh & 1, s = sh >> 1;
  int n = h * 16 + (lane & 15);         // output column 0..31
  int K = t + 16 * (lane >> 4);         // B-operand K within step
  int R = 32 * s + K;                   // global K-row of Waug
  float val = (R < 512) ? W[(R >> 5) * 1024 + n * 32 + (R & 31)]
                        : bias[n * 32 + (R - 512)];
  _Float16 vh = (_Float16)val;
  ws[q]              = vh;
  ws[HALF_ELEMS + q] = (_Float16)(val - (float)vh);
}

static __device__ __forceinline__ v16h splat16(_Float16 x) {
  v16h v;
#pragma unroll
  for (int t = 0; t < 16; ++t) v[t] = x;
  return v;
}

// --------------------------------------------------------------------------
// Fused edge-network kernel. Per wave iteration: 32 edges = two 16-row
// M-tiles sharing every B read. A-tile for step s: diag(bond[:,s]) * Nmat,
// synthesized as packed-f16 hi/lo: ah = bsh*nh (v_pk_mul_f16), al captures
// the full residual (product rounding + both quantization residuals) with a
// chain of three v_pk_fma_f16. 3 WMMAs per (M-tile, N-half) then deliver
// f32-grade accuracy (~2^-21 rel) entirely on the f16 matrix pipe.
// --------------------------------------------------------------------------
__global__ __launch_bounds__(WG_THREADS) void edge_net_wmma(
    const float* __restrict__ atom, const float* __restrict__ bond,
    const int* __restrict__ pair, const _Float16* __restrict__ ws,
    float* __restrict__ out, int ntiles)
{
  __shared__ __align__(16) _Float16 B_lds[2 * HALF_ELEMS];              // 69632 B
  __shared__ __align__(16) float    bond_lds[WAVES_PER_WG * 32 * BOND_PITCH];
  {
    const float4* src = (const float4*)ws;
    float4*       dst = (float4*)B_lds;
    for (int i = threadIdx.x; i < (2 * HALF_ELEMS) / 8; i += WG_THREADS)
      dst[i] = src[i];
  }
  __syncthreads();

  const int lane = threadIdx.x & 31;
  const int wave = threadIdx.x >> 5;
  const int hi   = lane >> 4;
  const int M    = lane & 15;
  // 16-bit A layout K bases for this lane (contiguous, matches B): K = kb0+t (t<8), K = kb1+(t-8)
  const int kb0 = 16 * hi;
  const int kb1 = 16 * hi + 8;
  float* bw = bond_lds + wave * (32 * BOND_PITCH);

  for (int tile = blockIdx.x * WAVES_PER_WG + wave; tile < ntiles;
       tile += gridDim.x * WAVES_PER_WG) {
    // ---- stage bond tile (32 rows x 16) into per-wave LDS, padded pitch ----
    {
      const float4* brow = (const float4*)(bond + (size_t)(tile * 32 + lane) * BD);
      float4 q0 = brow[0], q1 = brow[1], q2 = brow[2], q3 = brow[3];
      *(float4*)(bw + lane * BOND_PITCH + 0)  = q0;
      *(float4*)(bw + lane * BOND_PITCH + 4)  = q1;
      *(float4*)(bw + lane * BOND_PITCH + 8)  = q2;
      *(float4*)(bw + lane * BOND_PITCH + 12) = q3;
    }

    const int e0 = tile * 32 + M;      // M-tile 0 edge for this lane's row
    const int e1 = e0 + 16;            // M-tile 1 edge
    const int dst0 = pair[2 * e0], src0 = pair[2 * e0 + 1];
    const int dst1 = pair[2 * e1], src1 = pair[2 * e1 + 1];
    const float* n0r = atom + (size_t)src0 * D;
    const float* n1r = atom + (size_t)src1 * D;

    // Load neighbor rows and decompose once per tile: n = nh + nl (f16+f16).
    v16h nh0, nh1, nl0, nl1;
#pragma unroll
    for (int t = 0; t < 8; ++t) {
      float a0 = n0r[kb0 + t], b0 = n0r[kb1 + t];
      float a1 = n1r[kb0 + t], b1 = n1r[kb1 + t];
      _Float16 ha0 = (_Float16)a0, hb0 = (_Float16)b0;
      _Float16 ha1 = (_Float16)a1, hb1 = (_Float16)b1;
      nh0[t] = ha0;  nh0[8 + t] = hb0;
      nh1[t] = ha1;  nh1[8 + t] = hb1;
      nl0[t] = (_Float16)(a0 - (float)ha0);  nl0[8 + t] = (_Float16)(b0 - (float)hb0);
      nl1[t] = (_Float16)(a1 - (float)ha1);  nl1[8 + t] = (_Float16)(b1 - (float)hb1);
    }

    const float* br0 = bw + M * BOND_PITCH;
    const float* br1 = bw + (16 + M) * BOND_PITCH;

    v8f c00 = {}, c01 = {}, c10 = {}, c11 = {};

#pragma unroll 1
    for (int s = 0; s < 16; ++s) {
      const float bs0 = br0[s];
      const float bs1 = br1[s];
      const _Float16* base = B_lds + (2 * s) * 512 + lane * 16;
      v16h bh0 = *(const v16h*)(base);
      v16h bh1 = *(const v16h*)(base + 512);
      v16h bl0 = *(const v16h*)(base + HALF_ELEMS);
      v16h bl1 = *(const v16h*)(base + HALF_ELEMS + 512);

      // bond scalar hi/lo split
      const _Float16 bq0h = (_Float16)bs0;
      const _Float16 bq1h = (_Float16)bs1;
      const _Float16 bq0l = (_Float16)(bs0 - (float)bq0h);
      const _Float16 bq1l = (_Float16)(bs1 - (float)bq1h);
      const v16h vb0h = splat16(bq0h), vb0l = splat16(bq0l);
      const v16h vb1h = splat16(bq1h), vb1l = splat16(bq1l);

      // ah = bsh (x) nh            : 8x v_pk_mul_f16 per tile
      v16h ah0 = nh0 * bq0h;
      v16h ah1 = nh1 * bq1h;
      // al = fma(bsh,nl, fma(bsl,nh, fma(bsh,nh, -ah)))  : 3x8 v_pk_fma_f16
      v16h al0 = __builtin_elementwise_fma(vb0h, nh0, -ah0);
      v16h al1 = __builtin_elementwise_fma(vb1h, nh1, -ah1);
      al0 = __builtin_elementwise_fma(vb0l, nh0, al0);
      al1 = __builtin_elementwise_fma(vb1l, nh1, al1);
      al0 = __builtin_elementwise_fma(vb0h, nl0, al0);
      al1 = __builtin_elementwise_fma(vb1h, nl1, al1);

      c00 = __builtin_amdgcn_wmma_f32_16x16x32_f16(false, ah0, false, bh0, (short)0, c00, false, false);
      c01 = __builtin_amdgcn_wmma_f32_16x16x32_f16(false, ah0, false, bh1, (short)0, c01, false, false);
      c10 = __builtin_amdgcn_wmma_f32_16x16x32_f16(false, ah1, false, bh0, (short)0, c10, false, false);
      c11 = __builtin_amdgcn_wmma_f32_16x16x32_f16(false, ah1, false, bh1, (short)0, c11, false, false);
      c00 = __builtin_amdgcn_wmma_f32_16x16x32_f16(false, ah0, false, bl0, (short)0, c00, false, false);
      c01 = __builtin_amdgcn_wmma_f32_16x16x32_f16(false, ah0, false, bl1, (short)0, c01, false, false);
      c10 = __builtin_amdgcn_wmma_f32_16x16x32_f16(false, ah1, false, bl0, (short)0, c10, false, false);
      c11 = __builtin_amdgcn_wmma_f32_16x16x32_f16(false, ah1, false, bl1, (short)0, c11, false, false);
      c00 = __builtin_amdgcn_wmma_f32_16x16x32_f16(false, al0, false, bh0, (short)0, c00, false, false);
      c01 = __builtin_amdgcn_wmma_f32_16x16x32_f16(false, al0, false, bh1, (short)0, c01, false, false);
      c10 = __builtin_amdgcn_wmma_f32_16x16x32_f16(false, al1, false, bh0, (short)0, c10, false, false);
      c11 = __builtin_amdgcn_wmma_f32_16x16x32_f16(false, al1, false, bh1, (short)0, c11, false, false);
    }

    // ---- bias step (K rows 512..543): A = Nmat, i.e. ah = nh, al = nl ----
    {
      const _Float16* base = B_lds + 32 * 512 + lane * 16;
      v16h bh0 = *(const v16h*)(base);
      v16h bh1 = *(const v16h*)(base + 512);
      v16h bl0 = *(const v16h*)(base + HALF_ELEMS);
      v16h bl1 = *(const v16h*)(base + HALF_ELEMS + 512);

      c00 = __builtin_amdgcn_wmma_f32_16x16x32_f16(false, nh0, false, bh0, (short)0, c00, false, false);
      c01 = __builtin_amdgcn_wmma_f32_16x16x32_f16(false, nh0, false, bh1, (short)0, c01, false, false);
      c10 = __builtin_amdgcn_wmma_f32_16x16x32_f16(false, nh1, false, bh0, (short)0, c10, false, false);
      c11 = __builtin_amdgcn_wmma_f32_16x16x32_f16(false, nh1, false, bh1, (short)0, c11, false, false);
      c00 = __builtin_amdgcn_wmma_f32_16x16x32_f16(false, nh0, false, bl0, (short)0, c00, false, false);
      c01 = __builtin_amdgcn_wmma_f32_16x16x32_f16(false, nh0, false, bl1, (short)0, c01, false, false);
      c10 = __builtin_amdgcn_wmma_f32_16x16x32_f16(false, nh1, false, bl0, (short)0, c10, false, false);
      c11 = __builtin_amdgcn_wmma_f32_16x16x32_f16(false, nh1, false, bl1, (short)0, c11, false, false);
      c00 = __builtin_amdgcn_wmma_f32_16x16x32_f16(false, nl0, false, bh0, (short)0, c00, false, false);
      c01 = __builtin_amdgcn_wmma_f32_16x16x32_f16(false, nl0, false, bh1, (short)0, c01, false, false);
      c10 = __builtin_amdgcn_wmma_f32_16x16x32_f16(false, nl1, false, bh0, (short)0, c10, false, false);
      c11 = __builtin_amdgcn_wmma_f32_16x16x32_f16(false, nl1, false, bh1, (short)0, c11, false, false);
    }

    // C layout: VGPR r -> row M=r (lanes 0-15) / r+8 (lanes 16-31), col N = lane%16 (+16)
#pragma unroll
    for (int r = 0; r < 8; ++r) {
      int m  = r + 8 * hi;
      int d0 = __shfl(dst0, m, 32);
      int d1 = __shfl(dst1, m, 32);
      (void)__hip_atomic_fetch_add(&out[(size_t)d0 * D + M],      c00[r], __ATOMIC_RELAXED, __HIP_MEMORY_SCOPE_AGENT);
      (void)__hip_atomic_fetch_add(&out[(size_t)d0 * D + M + 16], c01[r], __ATOMIC_RELAXED, __HIP_MEMORY_SCOPE_AGENT);
      (void)__hip_atomic_fetch_add(&out[(size_t)d1 * D + M],      c10[r], __ATOMIC_RELAXED, __HIP_MEMORY_SCOPE_AGENT);
      (void)__hip_atomic_fetch_add(&out[(size_t)d1 * D + M + 16], c11[r], __ATOMIC_RELAXED, __HIP_MEMORY_SCOPE_AGENT);
    }
  }
}

// Scalar tail for E % 32 edges (unused for E = 400000, kept for generality).
__global__ void edge_tail(const float* __restrict__ atom, const float* __restrict__ bond,
                          const int* __restrict__ pair, const float* __restrict__ W,
                          const float* __restrict__ bias, float* __restrict__ out,
                          int estart, int E) {
  int gid = blockIdx.x * blockDim.x + threadIdx.x;
  int e = estart + gid / 32;
  int i = gid % 32;
  if (e >= E) return;
  int a0 = pair[2 * e], a1 = pair[2 * e + 1];
  const float* n = atom + (size_t)a1 * D;
  float acc = 0.f;
  for (int j = 0; j < 32; ++j) {
    float m = bias[i * 32 + j];
    for (int k = 0; k < 16; ++k) m += bond[(size_t)e * BD + k] * W[k * 1024 + i * 32 + j];
    acc += m * n[j];
  }
  (void)__hip_atomic_fetch_add(&out[(size_t)a0 * D + i], acc, __ATOMIC_RELAXED, __HIP_MEMORY_SCOPE_AGENT);
}

extern "C" void kernel_launch(void* const* d_in, const int* in_sizes, int n_in,
                              void* d_out, int out_size, void* d_ws, size_t ws_size,
                              hipStream_t stream) {
  const float* atom = (const float*)d_in[0];
  const float* bond = (const float*)d_in[1];
  const int*   pair = (const int*)d_in[2];
  const float* W    = (const float*)d_in[3];
  const float* bias = (const float*)d_in[4];
  float*     out = (float*)d_out;
  _Float16*  ws  = (_Float16*)d_ws;     // needs 2*HALF_ELEMS*2 = 69632 bytes

  const int E = in_sizes[1] / BD;

  hipMemsetAsync(d_out, 0, sizeof(float) * (size_t)out_size, stream);
  prep_waug<<<(HALF_ELEMS + 255) / 256, 256, 0, stream>>>(W, bias, ws);

  const int ntiles = E / 32;
  if (ntiles > 0) {
    int grid = (ntiles + WAVES_PER_WG - 1) / WAVES_PER_WG;
    if (grid > 1024) grid = 1024;
    edge_net_wmma<<<grid, WG_THREADS, 0, stream>>>(atom, bond, pair, ws, out, ntiles);
  }
  const int erem = E - ntiles * 32;
  if (erem > 0) {
    int thr = erem * 32;
    edge_tail<<<(thr + 127) / 128, 128, 0, stream>>>(atom, bond, pair, W, bias, out,
                                                     ntiles * 32, E);
  }
}
